// DynamicConv_84447646974141
// MI455X (gfx1250) — matrix operand order repaired and checked by reading, not compiled, not run on hardware
//
#include <hip/hip_runtime.h>

#define BB 4
#define CC 64
#define HH 64
#define WW 64

typedef __attribute__((ext_vector_type(16))) _Float16 v16h;
typedef __attribute__((ext_vector_type(8)))  _Float16 v8h;
typedef __attribute__((ext_vector_type(8)))  float    v8f;

__device__ __forceinline__ v8f wmma16(v16h a, v16h b, v8f c) {
    return __builtin_amdgcn_wmma_f32_16x16x32_f16(false, a, false, b, (short)0, c, false, false);
}

// ---------------- conversion kernels ----------------

// NCHW fp32 -> NHWC f16  (B,H,W,64)
__global__ void k_cvt_x_nhwc(const float* __restrict__ x, _Float16* __restrict__ xh) {
    int idx = blockIdx.x * blockDim.x + threadIdx.x;
    if (idx >= BB * HH * WW * CC) return;
    int c = idx & 63, xx = (idx >> 6) & 63, y = (idx >> 12) & 63, b = idx >> 18;
    xh[idx] = (_Float16)x[(((b * 64 + c) * 64 + y) * 64) + xx];
}

// c-permutation within each 32-channel block so an A fragment is one contiguous
// 32B load per lane: stored position p holds original c = p (identity map, contiguous k = 16h + s).
__device__ __forceinline__ int a_perm_c(int cpos) {
    int blk = cpos >> 5, p = cpos & 31;
    int group = p >> 4, within = p & 15, sub = within >> 3, off = within & 7;
    return blk * 32 + group * 16 + sub * 8 + off;
}

// (O,Ci,3,3) fp32 -> (O,9,Ci) f16, c permuted per 32-block
__global__ void k_cvt_w3x3(const float* __restrict__ w, _Float16* __restrict__ wh, int O, int Ci) {
    int idx = blockIdx.x * blockDim.x + threadIdx.x;
    int total = O * 9 * Ci;
    if (idx >= total) return;
    int cpos = idx % Ci, t = (idx / Ci) % 9, o = idx / (9 * Ci);
    int c = a_perm_c(cpos);
    wh[idx] = (_Float16)w[(size_t)(o * Ci + c) * 9 + t];
}

// (O,Ci) fp32 -> (O,Ci) f16, c permuted per 32-block
__global__ void k_cvt_w1x1(const float* __restrict__ w, _Float16* __restrict__ wh, int O, int Ci) {
    int idx = blockIdx.x * blockDim.x + threadIdx.x;
    if (idx >= O * Ci) return;
    int cpos = idx % Ci, o = idx / Ci;
    wh[idx] = (_Float16)w[(size_t)o * Ci + a_perm_c(cpos)];
}

__global__ void k_zero(float* __restrict__ p, int n) {
    int idx = blockIdx.x * blockDim.x + threadIdx.x;
    if (idx < n) p[idx] = 0.f;
}

// ---------------- WMMA implicit-GEMM 3x3 conv (+bias+ReLU), f16 NHWC in/out ----------------
// MBx2 register blocking: each wave computes a (16*MB)(M-out-ch) x 32(N-pixel) block.
// in : (B,H,W,CIN) f16    wT : (COUT,9,CIN) f16 (c-permuted)    out : (B,H,W,COUT) f16
template <int CIN, int COUT, int MB>
__global__ __launch_bounds__(128) void k_conv3x3(const _Float16* __restrict__ in,
                                                 const _Float16* __restrict__ wT,
                                                 const float* __restrict__ bias,
                                                 _Float16* __restrict__ out,
                                                 const _Float16* __restrict__ zbuf) {
    const int lane = threadIdx.x & 31;
    int tile = blockIdx.x * (blockDim.x >> 5) + (threadIdx.x >> 5);
    const int OT = COUT / (16 * MB);
    int ot = tile % OT;  tile /= OT;
    int xt = tile % 2;   tile /= 2;
    int y  = tile % HH;  tile /= HH;
    int b  = tile;                        // 0..3
    const int n = lane & 15;              // col for B/D, row for A
    const int s = lane >> 4;
    const int obase = ot * (16 * MB);
    const int xbase = xt * 32;

    v8f acc[MB][2];
    #pragma unroll
    for (int mi = 0; mi < MB; ++mi) { acc[mi][0] = (v8f){}; acc[mi][1] = (v8f){}; }

    #pragma unroll
    for (int ky = 0; ky < 3; ++ky) {
        int yy = y + ky - 1;
        if (yy < 0 || yy >= HH) continue;            // wave-uniform
        #pragma unroll
        for (int kx = 0; kx < 3; ++kx) {
            int xx0 = xbase + n + kx - 1;
            int xx1 = xx0 + 16;
            bool ok0 = (xx0 >= 0 && xx0 < WW);
            bool ok1 = (xx1 >= 0 && xx1 < WW);
            size_t rowb = (size_t)((b * HH + yy) * WW);
            // out-of-range lanes read a zeroed scratch region: no cndmasks,
            // EXEC stays all-ones for every WMMA.
            const _Float16* brow0 = ok0 ? in + (rowb + xx0) * CIN + (s << 4) : zbuf + (s << 4);
            const _Float16* brow1 = ok1 ? in + (rowb + xx1) * CIN + (s << 4) : zbuf + (s << 4);
            const _Float16* arow[MB];
            #pragma unroll
            for (int mi = 0; mi < MB; ++mi)
                arow[mi] = wT + ((size_t)(obase + mi * 16 + n) * 9 + (ky * 3 + kx)) * CIN + (s << 4);
            for (int cb = 0; cb < CIN; cb += 32) {
                v16h a[MB];
                #pragma unroll
                for (int mi = 0; mi < MB; ++mi) a[mi] = *(const v16h*)(arow[mi] + cb);
                v16h b0 = *(const v16h*)(brow0 + cb);
                v16h b1 = *(const v16h*)(brow1 + cb);
                #pragma unroll
                for (int mi = 0; mi < MB; ++mi) {
                    acc[mi][0] = wmma16(a[mi], b0, acc[mi][0]);
                    acc[mi][1] = wmma16(a[mi], b1, acc[mi][1]);
                }
            }
        }
    }
    // D: lane n holds column, rows m = r + 8s
    #pragma unroll
    for (int mi = 0; mi < MB; ++mi) {
        #pragma unroll
        for (int ni = 0; ni < 2; ++ni) {
            size_t opix = ((size_t)((b * HH + y) * WW + xbase + ni * 16 + n)) * COUT
                        + obase + mi * 16 + (s << 3);
            v8h ov;
            #pragma unroll
            for (int r = 0; r < 8; ++r) {
                float v = acc[mi][ni][r] + bias[obase + mi * 16 + (s << 3) + r];
                ov[r] = (_Float16)(v > 0.f ? v : 0.f);
            }
            *(v8h*)(out + opix) = ov;
        }
    }
}

// ---------------- WMMA 1x1 conv: 1024 -> 64 ch, +bias, fp32 NHWC out (2x2 blocked) ----------------
__global__ __launch_bounds__(128) void k_conv1x1(const _Float16* __restrict__ in,   // (B,H,W,1024)
                                                 const _Float16* __restrict__ w,    // (64,1024) permuted
                                                 const float* __restrict__ bias,
                                                 float* __restrict__ out) {         // (B,H,W,64)
    const int lane = threadIdx.x & 31;
    int tile = blockIdx.x * (blockDim.x >> 5) + (threadIdx.x >> 5);
    int ot = tile % 2;  tile /= 2;
    int xt = tile % 2;  tile /= 2;
    int y  = tile % HH; tile /= HH;
    int b  = tile;
    const int n = lane & 15, s = lane >> 4;
    const int obase = ot * 32, xbase = xt * 32;

    const _Float16* brow0 = in + ((size_t)((b * HH + y) * WW + xbase + n)) * 1024 + (s << 4);
    const _Float16* brow1 = brow0 + (size_t)16 * 1024;
    const _Float16* arow0 = w + (size_t)(obase + n) * 1024 + (s << 4);
    const _Float16* arow1 = arow0 + (size_t)16 * 1024;
    v8f acc00 = {}, acc01 = {}, acc10 = {}, acc11 = {};
    for (int cb = 0; cb < 1024; cb += 32) {
        v16h a0 = *(const v16h*)(arow0 + cb);
        v16h a1 = *(const v16h*)(arow1 + cb);
        v16h b0 = *(const v16h*)(brow0 + cb);
        v16h b1 = *(const v16h*)(brow1 + cb);
        acc00 = wmma16(a0, b0, acc00);
        acc01 = wmma16(a0, b1, acc01);
        acc10 = wmma16(a1, b0, acc10);
        acc11 = wmma16(a1, b1, acc11);
    }
    #pragma unroll
    for (int mi = 0; mi < 2; ++mi) {
        #pragma unroll
        for (int ni = 0; ni < 2; ++ni) {
            const v8f& a = mi == 0 ? (ni == 0 ? acc00 : acc01) : (ni == 0 ? acc10 : acc11);
            size_t opix = ((size_t)((b * HH + y) * WW + xbase + ni * 16 + n)) * CC
                        + obase + mi * 16 + (s << 3);
            #pragma unroll
            for (int r = 0; r < 8; ++r)
                out[opix + r] = a[r] + bias[obase + mi * 16 + (s << 3) + r];
        }
    }
}

// ---------------- tail kernels ----------------

// pooled[b][c] = mean over H*W of h32 (NHWC)
__global__ void k_pool(const float* __restrict__ h32, float* __restrict__ pooled) {
    __shared__ float ss[256];
    int b = blockIdx.x >> 6, c = blockIdx.x & 63;
    float sum = 0.f;
    for (int i = threadIdx.x; i < HH * WW; i += 256)
        sum += h32[((size_t)(b * HH * WW + i)) * CC + c];
    ss[threadIdx.x] = sum;
    __syncthreads();
    for (int k = 128; k > 0; k >>= 1) {
        if (threadIdx.x < (unsigned)k) ss[threadIdx.x] += ss[threadIdx.x + k];
        __syncthreads();
    }
    if (threadIdx.x == 0) pooled[b * CC + c] = ss[0] / (float)(HH * WW);
}

// dw[b][r] = pooled[b,:] . wd[r,:] + bd[r]   (r = c*9+p)
__global__ void k_dw(const float* __restrict__ pooled, const float* __restrict__ wd,
                     const float* __restrict__ bd, float* __restrict__ dwb) {
    int idx = blockIdx.x * blockDim.x + threadIdx.x;
    if (idx >= BB * 576) return;
    int b = idx / 576, r = idx % 576;
    float s = bd[r];
    for (int k = 0; k < CC; ++k) s += pooled[b * CC + k] * wd[r * CC + k];
    dwb[idx] = s;
}

// cols[i][a][c][x]: column sums of h over rows {0..62, 0..63, 1..63}
__global__ void k_cols(const float* __restrict__ h32, float* __restrict__ cols) {
    int idx = blockIdx.x * blockDim.x + threadIdx.x;
    if (idx >= BB * CC * WW) return;
    int x = idx & 63, c = (idx >> 6) & 63, a = idx >> 12;
    float full = 0.f, first = 0.f, last = 0.f;
    for (int y = 0; y < HH; ++y) {
        float v = h32[((size_t)((a * HH + y) * WW + x)) * CC + c];
        full += v;
        if (y == 0) first = v;
        if (y == HH - 1) last = v;
    }
    size_t base = ((size_t)(a * CC + c)) * WW + x;
    cols[0 * BB * CC * WW + base] = full - last;   // tap row i=0 -> rows 0..62
    cols[1 * BB * CC * WW + base] = full;          // i=1 -> rows 0..63
    cols[2 * BB * CC * WW + base] = full - first;  // i=2 -> rows 1..63
}

// xd[a][b][c][w] = sum_{i,j} dw[b,c,i*3+j] * cols[i][a][c][w+j-1]
__global__ void k_xd(const float* __restrict__ cols, const float* __restrict__ dwb,
                     float* __restrict__ xd) {
    int idx = blockIdx.x * blockDim.x + threadIdx.x;
    if (idx >= BB * BB * CC * WW) return;
    int w = idx & 63, c = (idx >> 6) & 63, b = (idx >> 12) & 3, a = idx >> 14;
    float s = 0.f;
    #pragma unroll
    for (int i = 0; i < 3; ++i) {
        #pragma unroll
        for (int j = 0; j < 3; ++j) {
            int wc = w + j - 1;
            if (wc >= 0 && wc < WW)
                s += dwb[b * 576 + c * 9 + i * 3 + j] *
                     cols[i * BB * CC * WW + ((size_t)(a * CC + c)) * WW + wc];
        }
    }
    xd[idx] = s;
}

// out_raw[a][o][c][w] = sum_b xd[a,b,c,w]*w4[o,b] + b4[o]
__global__ void k_out(const float* __restrict__ xd, const float* __restrict__ w4,
                      const float* __restrict__ b4, float* __restrict__ outr) {
    int idx = blockIdx.x * blockDim.x + threadIdx.x;
    if (idx >= BB * CC * CC * WW) return;
    int w = idx & 63, c = (idx >> 6) & 63, o = (idx >> 12) & 63, a = idx >> 18;
    float s = b4[o];
    #pragma unroll
    for (int b = 0; b < BB; ++b)
        s += xd[((size_t)((a * BB + b) * CC + c)) * WW + w] * w4[o * BB + b];
    outr[idx] = s;
}

// per-o mean/var over dims (a,c,w)
__global__ void k_stats(const float* __restrict__ outr, float* __restrict__ meanv,
                        float* __restrict__ varv) {
    __shared__ float ss[256], sq[256];
    int o = blockIdx.x;
    float s = 0.f, q = 0.f;
    for (int t = threadIdx.x; t < BB * CC * WW; t += 256) {
        int a = t >> 12, rem = t & 4095;
        float v = outr[(size_t)a * (CC * CC * WW) + (size_t)o * (CC * WW) + rem];
        s += v; q += v * v;
    }
    ss[threadIdx.x] = s; sq[threadIdx.x] = q;
    __syncthreads();
    for (int k = 128; k > 0; k >>= 1) {
        if (threadIdx.x < (unsigned)k) {
            ss[threadIdx.x] += ss[threadIdx.x + k];
            sq[threadIdx.x] += sq[threadIdx.x + k];
        }
        __syncthreads();
    }
    if (threadIdx.x == 0) {
        float m = ss[0] / 16384.f;
        meanv[o] = m;
        varv[o] = sq[0] / 16384.f - m * m;
    }
}

__global__ void k_bn(const float* __restrict__ outr, const float* __restrict__ meanv,
                     const float* __restrict__ varv, const float* __restrict__ gamma,
                     const float* __restrict__ beta, float* __restrict__ out) {
    int idx = blockIdx.x * blockDim.x + threadIdx.x;
    if (idx >= BB * CC * CC * WW) return;
    int o = (idx >> 12) & 63;
    out[idx] = (outr[idx] - meanv[o]) * rsqrtf(varv[o] + 1e-5f) * gamma[o] + beta[o];
}

// ---------------- launcher ----------------
extern "C" void kernel_launch(void* const* d_in, const int* in_sizes, int n_in,
                              void* d_out, int out_size, void* d_ws, size_t ws_size,
                              hipStream_t stream) {
    (void)in_sizes; (void)n_in; (void)out_size; (void)ws_size;
    const float* x  = (const float*)d_in[0];
    const float* w1 = (const float*)d_in[1];
    const float* b1 = (const float*)d_in[2];
    const float* w2 = (const float*)d_in[3];
    const float* b2 = (const float*)d_in[4];
    const float* w3 = (const float*)d_in[5];
    const float* b3 = (const float*)d_in[6];
    const float* wd = (const float*)d_in[7];
    const float* bd = (const float*)d_in[8];
    const float* w4 = (const float*)d_in[9];
    const float* b4 = (const float*)d_in[10];
    const float* gamma = (const float*)d_in[11];
    const float* beta  = (const float*)d_in[12];
    float* out = (float*)d_out;

    char* p = (char*)d_ws;
    auto alloc = [&](size_t bytes) {
        void* r = (void*)p;
        p += (bytes + 255) & ~(size_t)255;
        return r;
    };
    _Float16* xh   = (_Float16*)alloc((size_t)BB * HH * WW * CC * 2);          // 2 MB
    _Float16* w1h  = (_Float16*)alloc((size_t)256 * 9 * 64 * 2);
    _Float16* w2h  = (_Float16*)alloc((size_t)1024 * 9 * 256 * 2);
    _Float16* w3h  = (_Float16*)alloc((size_t)64 * 1024 * 2);
    _Float16* h1h  = (_Float16*)alloc((size_t)BB * HH * WW * 256 * 2);         // 8 MB
    _Float16* h2h  = (_Float16*)alloc((size_t)BB * HH * WW * 1024 * 2);        // 32 MB
    float*    h32  = (float*)alloc((size_t)BB * HH * WW * CC * 4);             // 4 MB
    float*    pooled = (float*)alloc((size_t)BB * CC * 4);
    float*    dwb    = (float*)alloc((size_t)BB * 576 * 4);
    float*    cols   = (float*)alloc((size_t)3 * BB * CC * WW * 4);
    float*    xd     = (float*)alloc((size_t)BB * BB * CC * WW * 4);
    float*    outr   = (float*)alloc((size_t)BB * CC * CC * WW * 4);           // 4 MB
    float*    meanv  = (float*)alloc(64 * 4);
    float*    varv   = (float*)alloc(64 * 4);
    float*    zbuf   = (float*)alloc(2048);                                    // zero halo source

    // conversions + zero scratch
    k_zero<<<2, 256, 0, stream>>>(zbuf, 512);
    k_cvt_x_nhwc<<<(BB * HH * WW * CC + 255) / 256, 256, 0, stream>>>(x, xh);
    k_cvt_w3x3<<<(256 * 9 * 64 + 255) / 256, 256, 0, stream>>>(w1, w1h, 256, 64);
    k_cvt_w3x3<<<(1024 * 9 * 256 + 255) / 256, 256, 0, stream>>>(w2, w2h, 1024, 256);
    k_cvt_w1x1<<<(64 * 1024 + 255) / 256, 256, 0, stream>>>(w3, w3h, 64, 1024);

    // conv1 (2x2 blocking): waves = 4*64*2*(256/32)=4096, 4 waves/block
    k_conv3x3<64, 256, 2><<<4096 / 4, 128, 0, stream>>>(xh, w1h, b1, h1h, (const _Float16*)zbuf);
    // conv2 (4x2 blocking): waves = 4*64*2*(1024/64)=8192
    k_conv3x3<256, 1024, 4><<<8192 / 4, 128, 0, stream>>>(h1h, w2h, b2, h2h, (const _Float16*)zbuf);
    // conv3 (1x1, 2x2 blocking): waves = 4*64*2*2 = 1024
    k_conv1x1<<<1024 / 4, 128, 0, stream>>>(h2h, w3h, b3, h32);

    // tail
    k_pool<<<BB * CC, 256, 0, stream>>>(h32, pooled);
    k_dw<<<(BB * 576 + 255) / 256, 256, 0, stream>>>(pooled, wd, bd, dwb);
    k_cols<<<(BB * CC * WW + 255) / 256, 256, 0, stream>>>(h32, cols);
    k_xd<<<(BB * BB * CC * WW + 255) / 256, 256, 0, stream>>>(cols, dwb, xd);
    k_out<<<(BB * CC * CC * WW + 255) / 256, 256, 0, stream>>>(xd, w4, b4, outr);
    k_stats<<<64, 256, 0, stream>>>(outr, meanv, varv);
    k_bn<<<(BB * CC * CC * WW + 255) / 256, 256, 0, stream>>>(outr, meanv, varv, gamma, beta, out);
}
